// MatchingNetwork_9320079032859
// MI455X (gfx1250) — compile-verified, matrix operand order checked by reading, not run on hardware
//
#include <hip/hip_runtime.h>
#include <hip/hip_bf16.h>

typedef __attribute__((ext_vector_type(16))) _Float16 v16h;
typedef __attribute__((ext_vector_type(8)))  _Float16 v8h;
typedef __attribute__((ext_vector_type(8)))  float    v8f;
typedef __attribute__((ext_vector_type(4)))  unsigned int v4u;
typedef __attribute__((ext_vector_type(8)))  int v8i;
typedef __attribute__((ext_vector_type(4)))  int v4i;

#define HD 128
#define ED 64
#define LSEQ 1024
#define NSUP 65536
#define LQ 1024
#define NPART 8

#if defined(__has_builtin)
#if __has_builtin(__builtin_amdgcn_tensor_load_to_lds) && \
    __has_builtin(__builtin_amdgcn_s_wait_tensorcnt)
#define USE_TDM 1
#endif
#endif

// ---------------------------------------------------------------------------
// WMMA helpers
// ---------------------------------------------------------------------------
__device__ __forceinline__ v8f wmma16(v16h a, v16h b, v8f c) {
  return __builtin_amdgcn_wmma_f32_16x16x32_f16(
      false, a, false, b, (short)0, c, false, false);
}

__device__ __forceinline__ v8f vzero8() {
  v8f z;
#pragma unroll
  for (int i = 0; i < 8; ++i) z[i] = 0.f;
  return z;
}

// Load a 16x32 f16 fragment (A from row-major [M][ld], or B from row-major
// Bt[N][ld]). Lane l holds row (row0 + l%16); k-halves at kbase and kbase+16.
__device__ __forceinline__ v16h ld_frag16(const _Float16* base, int ld,
                                          int row0, int col0, int lane) {
  const int m  = lane & 15;
  const int kb = (lane >> 4) << 3;  // 0 or 8
  const _Float16* p = base + (size_t)(row0 + m) * ld + col0 + kb;
  v8h lo = *(const v8h*)(p);
  v8h hi = *(const v8h*)(p + 16);
  v16h f;
#pragma unroll
  for (int i = 0; i < 8; ++i) { f[i] = lo[i]; f[8 + i] = hi[i]; }
  return f;
}

// Store a 16x16 f32 C/D tile to row-major [ld] floats, scaled.
__device__ __forceinline__ void st_ctile(float* base, int ld, int row0,
                                         int col0, int lane, v8f c, float s) {
  const int n  = lane & 15;
  const int mb = (lane >> 4) << 3;
#pragma unroll
  for (int r = 0; r < 8; ++r)
    base[(size_t)(row0 + mb + r) * ld + col0 + n] = c[r] * s;
}

// ---------------------------------------------------------------------------
// Tensor Data Mover: 2D tile load (global -> LDS) with optional row padding.
// D# packed per CDNA5 ISA 8.3/8.4 (group0: count/lds/global/type;
// group1: data_size|pad cfg, tensor dims, tile dims, dim0 stride).
// ---------------------------------------------------------------------------
__device__ __forceinline__ void tdm_load_2d(
    unsigned long long gaddr, unsigned int lds_addr, unsigned int cfg_word,
    unsigned int tensor_d0, unsigned int tensor_d1,
    unsigned int tile_d0, unsigned int tile_d1, unsigned long long stride0) {
#ifdef USE_TDM
  v4u g0;
  g0[0] = 1u;                                            // count=1, user mode
  g0[1] = lds_addr;                                      // lds_addr [63:32]
  g0[2] = (unsigned int)(gaddr & 0xffffffffull);         // global_addr lo
  g0[3] = (unsigned int)((gaddr >> 32) & 0x1ffffffull) | (2u << 30);  // type=2
  v8i g1;
  g1[0] = (int)cfg_word;                                 // mask|data_size|pad
  g1[1] = (int)((tensor_d0 & 0xffffu) << 16);            // tensor_dim0 lo16
  g1[2] = (int)((tensor_d0 >> 16) | ((tensor_d1 & 0xffffu) << 16));
  g1[3] = (int)((tensor_d1 >> 16) | ((tile_d0 & 0xffffu) << 16));
  g1[4] = (int)(tile_d1 & 0xffffu);                      // tile_dim1; dim2=0
  g1[5] = (int)(unsigned int)(stride0 & 0xffffffffull);  // dim0 stride lo
  g1[6] = (int)(unsigned int)((stride0 >> 32) & 0xffffu);
  g1[7] = 0;
  v4i z4 = {0, 0, 0, 0};
#if __clang_major__ >= 23
  v8i z8 = {0, 0, 0, 0, 0, 0, 0, 0};
  __builtin_amdgcn_tensor_load_to_lds(g0, g1, z4, z4, z8, 0);
#else
  __builtin_amdgcn_tensor_load_to_lds(g0, g1, z4, z4, 0);
#endif
#else
  (void)gaddr; (void)lds_addr; (void)cfg_word; (void)tensor_d0;
  (void)tensor_d1; (void)tile_d0; (void)tile_d1; (void)stride0;
#endif
}

// cfg: data_size=1 (2B) | pad_enable | pad_interval<<22 | pad_amount<<25
#define TDM_CFG(interval, amount) \
  ((1u << 16) | (1u << 20) | ((unsigned)(interval) << 22) | ((unsigned)(amount) << 25))

// ---------------------------------------------------------------------------
// Embedding gathers
// ---------------------------------------------------------------------------
__global__ void embed_kernel(const int* __restrict__ ids,
                             const float* __restrict__ emb,
                             _Float16* __restrict__ out, int n, int ed) {
  int i = blockIdx.x * blockDim.x + threadIdx.x;
  if (i >= n * ed) return;
  int pos = i / ed, c = i - pos * ed;
  out[i] = (_Float16)emb[ids[pos] * ed + c];
}

// Transposed gather: outT[c][pos] = emb[ids[pos]][c]  (outT is [ed][n])
__global__ void embed_t_kernel(const int* __restrict__ ids,
                               const float* __restrict__ emb,
                               _Float16* __restrict__ outT, int n, int ed) {
  int i = blockIdx.x * blockDim.x + threadIdx.x;
  if (i >= n * ed) return;
  int c = i / n, pos = i - c * n;
  outT[i] = (_Float16)emb[ids[pos] * ed + c];
}

// ---------------------------------------------------------------------------
// Fused conv1d(k=3,pad=1) + bias + ReLU + LayerNorm.
// Block: 256 threads (8 waves), 128 positions x 128 out-channels.
// Dynamic LDS: aT f16[130][cin+8] | bW f16[3][128][cin+8] | cT f32[128][132]
// ---------------------------------------------------------------------------
__global__ __launch_bounds__(256) void conv_ln_kernel(
    const _Float16* __restrict__ in_act, _Float16* __restrict__ out_act,
    float* __restrict__ out_f32, const float* __restrict__ w,
    const float* __restrict__ bias, const float* __restrict__ lng,
    const float* __restrict__ lnb, int cin, int seq_len) {
  extern __shared__ char smem[];
  const int AP  = cin + 8;
  const int CTP = 132;
  _Float16* aT = (_Float16*)smem;
  _Float16* bW = aT + 130 * AP;
  float*    cT = (float*)(bW + 3 * 128 * AP);

  const int tid  = threadIdx.x;
  const int wave = tid >> 5;
  const int lane = tid & 31;
  const int t0   = blockIdx.x * 128;
  const int seq_lo = (t0 / seq_len) * seq_len;
  const int seq_hi = seq_lo + seq_len;

  for (int i = tid; i < 130 * cin; i += 256) {
    int r = i / cin, c = i - r * cin;
    int p = t0 - 1 + r;
    _Float16 v = (_Float16)0.f;
    if (p >= seq_lo && p < seq_hi) v = in_act[(size_t)p * cin + c];
    aT[r * AP + c] = v;
  }
  for (int i = tid; i < 3 * 128 * cin; i += 256) {
    int dk  = i / (128 * cin);
    int rem = i - dk * 128 * cin;
    int o = rem / cin, ci = rem - o * cin;
    bW[(dk * 128 + o) * AP + ci] = (_Float16)w[(o * cin + ci) * 3 + dk];
  }
  __syncthreads();

  const int mrow = wave * 16;
  v8f acc[8];
#pragma unroll
  for (int n = 0; n < 8; ++n) acc[n] = vzero8();

  for (int dk = 0; dk < 3; ++dk) {
    const _Float16* bWd = bW + dk * 128 * AP;
    for (int kk = 0; kk < cin; kk += 32) {
      v16h a = ld_frag16(aT, AP, mrow + dk, kk, lane);
#pragma unroll
      for (int n = 0; n < 8; ++n) {
        v16h b = ld_frag16(bWd, AP, n * 16, kk, lane);
        acc[n] = wmma16(a, b, acc[n]);
      }
    }
  }
#pragma unroll
  for (int n = 0; n < 8; ++n) st_ctile(cT, CTP, mrow, n * 16, lane, acc[n], 1.f);
  __syncthreads();

  if (tid < 128) {
    const int m = tid;
    float s1 = 0.f, s2 = 0.f;
    for (int o = 0; o < 128; ++o) {
      float x = fmaxf(cT[m * CTP + o] + bias[o], 0.f);
      cT[m * CTP + o] = x;
      s1 += x; s2 += x * x;
    }
    const float mean = s1 * (1.f / 128.f);
    const float var  = fmaxf(s2 * (1.f / 128.f) - mean * mean, 0.f);
    const float inv  = rsqrtf(var + 1e-5f);
    const size_t p = (size_t)(t0 + m) * 128;
    for (int o = 0; o < 128; ++o) {
      float y = (cT[m * CTP + o] - mean) * inv * lng[o] + lnb[o];
      out_act[p + o] = (_Float16)y;
      if (out_f32) out_f32[p + o] = y;
    }
  }
}

// ---------------------------------------------------------------------------
// Flash attention partial with TDM-staged K/V chunks (double buffered).
// Block: 256 threads, 64 query rows, one key partition.
// Dynamic LDS: Qs[64][136] | Ks[2][64][136] | Vt[2][128][72] |
//              Ss[64][68]f32 | Ps[64][72] | mrow/lrow/srow[64]f32
// ---------------------------------------------------------------------------
#define QLD 136
#define KLD2 136
#define VLD 72
#define SLD 68
#define PLD 72
__global__ __launch_bounds__(256) void flash_attn_kernel(
    const _Float16* __restrict__ qf,     // [LQ][128]
    const _Float16* __restrict__ keys,   // [NSUP][128]
    const _Float16* __restrict__ valsT,  // [128][NSUP] (transposed values)
    float* __restrict__ part_acc, float* __restrict__ part_m,
    float* __restrict__ part_l, int nk_per_part) {
  extern __shared__ char smem[];
  _Float16* Qs   = (_Float16*)smem;                 // 64*136
  _Float16* Ks   = Qs + 64 * QLD;                   // 2*64*136
  _Float16* Vt   = Ks + 2 * 64 * KLD2;              // 2*128*72
  float*    Ss   = (float*)(Vt + 2 * 128 * VLD);    // 64*68
  _Float16* Ps   = (_Float16*)(Ss + 64 * SLD);      // 64*72
  float*    mrow = (float*)(Ps + 64 * PLD);
  float*    lrow = mrow + 64;
  float*    srow = lrow + 64;

  const int tid  = threadIdx.x;
  const int wave = tid >> 5;
  const int lane = tid & 31;
  const int qbase  = blockIdx.x * 64;
  const int kbase0 = blockIdx.y * nk_per_part;
  const int nchunks = nk_per_part / 64;

  for (int i = tid; i < 64 * 128; i += 256) {
    int r = i >> 7, c = i & 127;
    Qs[r * QLD + c] = qf[(size_t)(qbase + r) * 128 + c];
  }
  if (tid < 64) { mrow[tid] = -1e30f; lrow[tid] = 0.f; }

#ifdef USE_TDM
  // Issue chunk 0 (one wave issues; TDM is a wave-level DMA op).
  if (tid < 32) {
    const int kb = kbase0;
    tdm_load_2d((unsigned long long)(uintptr_t)(keys + (size_t)kb * 128),
                (unsigned int)(uintptr_t)(Ks), TDM_CFG(5, 3),
                128u, 64u, 128u, 64u, 128ull);
    tdm_load_2d((unsigned long long)(uintptr_t)(valsT + kb),
                (unsigned int)(uintptr_t)(Vt), TDM_CFG(4, 3),
                64u, 128u, 64u, 128u, (unsigned long long)NSUP);
  }
#endif
  __syncthreads();

  const int amt = (wave & 3) * 16;   // acc M tile
  const int acb = (wave >> 2) * 64;  // acc channel base (4 N tiles)
  v8f acc[4];
#pragma unroll
  for (int n = 0; n < 4; ++n) acc[n] = vzero8();

  const float sm_scale = 0.08838834764831845f;  // 1/sqrt(128)

  for (int ci = 0; ci < nchunks; ++ci) {
    const int bb = ci & 1;
#ifdef USE_TDM
    if (tid < 32) {
      if (ci + 1 < nchunks) {
        const int kb = kbase0 + (ci + 1) * 64;
        const int nb = (ci + 1) & 1;
        tdm_load_2d((unsigned long long)(uintptr_t)(keys + (size_t)kb * 128),
                    (unsigned int)(uintptr_t)(Ks + nb * 64 * KLD2),
                    TDM_CFG(5, 3), 128u, 64u, 128u, 64u, 128ull);
        tdm_load_2d((unsigned long long)(uintptr_t)(valsT + kb),
                    (unsigned int)(uintptr_t)(Vt + nb * 128 * VLD),
                    TDM_CFG(4, 3), 64u, 128u, 64u, 128u,
                    (unsigned long long)NSUP);
        __builtin_amdgcn_s_wait_tensorcnt(2);  // current chunk landed
      } else {
        __builtin_amdgcn_s_wait_tensorcnt(0);
      }
    }
    __syncthreads();
#else
    // Fallback: manual staging of chunk ci into buffer bb.
    {
      const int kb = kbase0 + ci * 64;
      for (int i = tid; i < 64 * 128; i += 256) {
        int r = i >> 7, d = i & 127;
        Ks[bb * 64 * KLD2 + r * KLD2 + d] = keys[(size_t)(kb + r) * 128 + d];
      }
      for (int i = tid; i < 128 * 64; i += 256) {
        int c = i >> 6, k = i & 63;
        Vt[bb * 128 * VLD + c * VLD + k] = valsT[(size_t)c * NSUP + kb + k];
      }
    }
    __syncthreads();
#endif
    const _Float16* KsB = Ks + bb * 64 * KLD2;
    const _Float16* VtB = Vt + bb * 128 * VLD;

    // S = scale * Q @ Kchunk^T : wave -> tiles (wave>>1, (wave&1)*2 + {0,1})
    {
      const int smt = (wave >> 1) * 16;
      const int snb = (wave & 1) * 2;
      v8f s0 = vzero8(), s1 = vzero8();
      for (int kk = 0; kk < 128; kk += 32) {
        v16h a  = ld_frag16(Qs, QLD, smt, kk, lane);
        v16h b0 = ld_frag16(KsB, KLD2, snb * 16, kk, lane);
        v16h b1 = ld_frag16(KsB, KLD2, (snb + 1) * 16, kk, lane);
        s0 = wmma16(a, b0, s0);
        s1 = wmma16(a, b1, s1);
      }
      st_ctile(Ss, SLD, smt, snb * 16, lane, s0, sm_scale);
      st_ctile(Ss, SLD, smt, (snb + 1) * 16, lane, s1, sm_scale);
    }
    __syncthreads();

    // Online softmax per row.
    if (tid < 64) {
      const int r = tid;
      float mo = mrow[r], mc = mo;
      for (int j = 0; j < 64; ++j) mc = fmaxf(mc, Ss[r * SLD + j]);
      float alpha = __expf(mo - mc);
      float lsum = 0.f;
      for (int j = 0; j < 64; ++j) {
        float e = __expf(Ss[r * SLD + j] - mc);
        Ps[r * PLD + j] = (_Float16)e;
        lsum += e;
      }
      lrow[r] = lrow[r] * alpha + lsum;
      mrow[r] = mc;
      srow[r] = alpha;
    }
    __syncthreads();

    // Rescale acc and accumulate P @ Vchunk.
    {
      const int mb = (lane >> 4) << 3;
      float sl[8];
#pragma unroll
      for (int r = 0; r < 8; ++r) sl[r] = srow[amt + mb + r];
#pragma unroll
      for (int n = 0; n < 4; ++n)
#pragma unroll
        for (int r = 0; r < 8; ++r) acc[n][r] *= sl[r];

      for (int kk = 0; kk < 64; kk += 32) {
        v16h a = ld_frag16(Ps, PLD, amt, kk, lane);
#pragma unroll
        for (int n = 0; n < 4; ++n) {
          v16h b = ld_frag16(VtB, VLD, acb + n * 16, kk, lane);
          acc[n] = wmma16(a, b, acc[n]);
        }
      }
    }
    __syncthreads();
  }

  // Write unnormalized partials + (m, l).
  {
    const int mb = (lane >> 4) << 3;
    const int nn = lane & 15;
    float* pb = part_acc + (size_t)blockIdx.y * LQ * 128;
#pragma unroll
    for (int n = 0; n < 4; ++n)
#pragma unroll
      for (int r = 0; r < 8; ++r)
        pb[(size_t)(qbase + amt + mb + r) * 128 + acb + n * 16 + nn] = acc[n][r];
    if (tid < 64) {
      part_m[blockIdx.y * LQ + qbase + tid] = mrow[tid];
      part_l[blockIdx.y * LQ + qbase + tid] = lrow[tid];
    }
  }
}

// Merge NPART flash partials -> attended [LQ][128] f32.
__global__ void attn_combine_kernel(const float* __restrict__ part_acc,
                                    const float* __restrict__ part_m,
                                    const float* __restrict__ part_l,
                                    float* __restrict__ attended) {
  const int row = blockIdx.x;
  const int c   = threadIdx.x;
  float mstar = -1e30f;
  for (int p = 0; p < NPART; ++p) mstar = fmaxf(mstar, part_m[p * LQ + row]);
  float l = 0.f, s = 0.f;
  for (int p = 0; p < NPART; ++p) {
    float e = __expf(part_m[p * LQ + row] - mstar);
    l += part_l[p * LQ + row] * e;
    s += part_acc[((size_t)p * LQ + row) * 128 + c] * e;
  }
  attended[(size_t)row * 128 + c] = s / l;
}

// ---------------------------------------------------------------------------
// Decoder: h = LN(relu([q,att] @ w1^T + b1)); out = h @ w2^T + b2
// ---------------------------------------------------------------------------
__global__ __launch_bounds__(256) void decoder_kernel(
    const float* __restrict__ qf32, const float* __restrict__ att,
    const float* __restrict__ w1, const float* __restrict__ b1,
    const float* __restrict__ lng, const float* __restrict__ lnb,
    const float* __restrict__ w2, const float* __restrict__ b2,
    float* __restrict__ out) {
  extern __shared__ char smem[];
  const int KLD = 264, HLD = 132;
  _Float16* comb = (_Float16*)smem;
  _Float16* w1t  = comb + 128 * KLD;
  float*    H    = (float*)(w1t + 128 * KLD);

  const int tid  = threadIdx.x;
  const int wave = tid >> 5;
  const int lane = tid & 31;
  const int p0   = blockIdx.x * 128;

  for (int i = tid; i < 128 * 256; i += 256) {
    int m = i >> 8, c = i & 255;
    float v = (c < 128) ? qf32[(size_t)(p0 + m) * 128 + c]
                        : att[(size_t)(p0 + m) * 128 + (c - 128)];
    comb[m * KLD + c] = (_Float16)v;
  }
  for (int i = tid; i < 128 * 256; i += 256) {
    int o = i >> 8, c = i & 255;
    w1t[o * KLD + c] = (_Float16)w1[i];
  }
  __syncthreads();

  const int mrow = wave * 16;
  v8f acc[8];
#pragma unroll
  for (int n = 0; n < 8; ++n) acc[n] = vzero8();
  for (int kk = 0; kk < 256; kk += 32) {
    v16h a = ld_frag16(comb, KLD, mrow, kk, lane);
#pragma unroll
    for (int n = 0; n < 8; ++n) {
      v16h b = ld_frag16(w1t, KLD, n * 16, kk, lane);
      acc[n] = wmma16(a, b, acc[n]);
    }
  }
#pragma unroll
  for (int n = 0; n < 8; ++n) st_ctile(H, HLD, mrow, n * 16, lane, acc[n], 1.f);
  __syncthreads();

  if (tid < 128) {
    const int m = tid;
    float s1 = 0.f, s2 = 0.f;
    for (int o = 0; o < 128; ++o) {
      float x = fmaxf(H[m * HLD + o] + b1[o], 0.f);
      H[m * HLD + o] = x;
      s1 += x; s2 += x * x;
    }
    const float mean = s1 * (1.f / 128.f);
    const float var  = fmaxf(s2 * (1.f / 128.f) - mean * mean, 0.f);
    const float inv  = rsqrtf(var + 1e-5f);
    float h[10];
    for (int j = 0; j < 10; ++j) h[j] = b2[j];
    for (int o = 0; o < 128; ++o) {
      float y = (H[m * HLD + o] - mean) * inv * lng[o] + lnb[o];
      for (int j = 0; j < 10; ++j) h[j] += y * w2[j * 128 + o];
    }
    for (int j = 0; j < 10; ++j) out[(size_t)(p0 + m) * 10 + j] = h[j];
  }
}

// ---------------------------------------------------------------------------
// Host orchestration
// ---------------------------------------------------------------------------
extern "C" void kernel_launch(void* const* d_in, const int* in_sizes, int n_in,
                              void* d_out, int out_size, void* d_ws, size_t ws_size,
                              hipStream_t stream) {
  const int*   sup_in  = (const int*)d_in[0];
  const int*   sup_out = (const int*)d_in[1];
  const int*   q_in    = (const int*)d_in[2];
  const float* qe_emb = (const float*)d_in[3];
  const float* qe_c1w = (const float*)d_in[4];  const float* qe_c1b = (const float*)d_in[5];
  const float* qe_c2w = (const float*)d_in[6];  const float* qe_c2b = (const float*)d_in[7];
  const float* qe_c3w = (const float*)d_in[8];  const float* qe_c3b = (const float*)d_in[9];
  const float* qe_n1g = (const float*)d_in[10]; const float* qe_n1b = (const float*)d_in[11];
  const float* qe_n2g = (const float*)d_in[12]; const float* qe_n2b = (const float*)d_in[13];
  const float* qe_n3g = (const float*)d_in[14]; const float* qe_n3b = (const float*)d_in[15];
  const float* se_emb = (const float*)d_in[16];
  const float* se_c1w = (const float*)d_in[17]; const float* se_c1b = (const float*)d_in[18];
  const float* se_c2w = (const float*)d_in[19]; const float* se_c2b = (const float*)d_in[20];
  const float* se_c3w = (const float*)d_in[21]; const float* se_c3b = (const float*)d_in[22];
  const float* se_n1g = (const float*)d_in[23]; const float* se_n1b = (const float*)d_in[24];
  const float* se_n2g = (const float*)d_in[25]; const float* se_n2b = (const float*)d_in[26];
  const float* se_n3g = (const float*)d_in[27]; const float* se_n3b = (const float*)d_in[28];
  const float* out_emb = (const float*)d_in[29];
  const float* dec_w1 = (const float*)d_in[30]; const float* dec_b1 = (const float*)d_in[31];
  const float* dec_lng = (const float*)d_in[32]; const float* dec_lnb = (const float*)d_in[33];
  const float* dec_w2 = (const float*)d_in[34]; const float* dec_b2 = (const float*)d_in[35];

  size_t off = 0;
  auto carve = [&](size_t bytes) {
    void* p = (char*)d_ws + off;
    off += (bytes + 255) & ~(size_t)255;
    return p;
  };
  _Float16* s_a   = (_Float16*)carve((size_t)NSUP * 128 * 2);
  _Float16* s_b   = (_Float16*)carve((size_t)NSUP * 128 * 2);
  _Float16* valsT = (_Float16*)carve((size_t)NSUP * 128 * 2);
  _Float16* q_a   = (_Float16*)carve((size_t)LQ * 128 * 2);
  _Float16* q_b   = (_Float16*)carve((size_t)LQ * 128 * 2);
  float* q32      = (float*)carve((size_t)LQ * 128 * 4);
  float* att      = (float*)carve((size_t)LQ * 128 * 4);
  float* pacc     = (float*)carve((size_t)NPART * LQ * 128 * 4);
  float* pm       = (float*)carve((size_t)NPART * LQ * 4);
  float* pl       = (float*)carve((size_t)NPART * LQ * 4);
  (void)ws_size; (void)n_in; (void)in_sizes; (void)out_size;

  const int smem64  = 130 * 72 * 2  + 3 * 128 * 72 * 2  + 128 * 132 * 4;
  const int smem128 = 130 * 136 * 2 + 3 * 128 * 136 * 2 + 128 * 132 * 4;
  const int smemdec = 128 * 264 * 2 * 2 + 128 * 132 * 4;
  const int smemfa  = 64 * QLD * 2 + 2 * 64 * KLD2 * 2 + 2 * 128 * VLD * 2 +
                      64 * SLD * 4 + 64 * PLD * 2 + 3 * 64 * 4;
  hipFuncSetAttribute((const void*)conv_ln_kernel,
                      hipFuncAttributeMaxDynamicSharedMemorySize, smem128);
  hipFuncSetAttribute((const void*)decoder_kernel,
                      hipFuncAttributeMaxDynamicSharedMemorySize, smemdec);
  hipFuncSetAttribute((const void*)flash_attn_kernel,
                      hipFuncAttributeMaxDynamicSharedMemorySize, smemfa);

  // ---- Support encoder ----
  embed_kernel<<<(NSUP * ED + 255) / 256, 256, 0, stream>>>(sup_in, se_emb, s_a, NSUP, ED);
  conv_ln_kernel<<<NSUP / 128, 256, smem64, stream>>>(
      s_a, s_b, nullptr, se_c1w, se_c1b, se_n1g, se_n1b, 64, LSEQ);
  conv_ln_kernel<<<NSUP / 128, 256, smem128, stream>>>(
      s_b, s_a, nullptr, se_c2w, se_c2b, se_n2g, se_n2b, 128, LSEQ);
  conv_ln_kernel<<<NSUP / 128, 256, smem128, stream>>>(
      s_a, s_b, nullptr, se_c3w, se_c3b, se_n3g, se_n3b, 128, LSEQ);  // keys

  // ---- Query encoder ----
  embed_kernel<<<(LQ * ED + 255) / 256, 256, 0, stream>>>(q_in, qe_emb, q_a, LQ, ED);
  conv_ln_kernel<<<LQ / 128, 256, smem64, stream>>>(
      q_a, q_b, nullptr, qe_c1w, qe_c1b, qe_n1g, qe_n1b, 64, LSEQ);
  conv_ln_kernel<<<LQ / 128, 256, smem128, stream>>>(
      q_b, q_a, nullptr, qe_c2w, qe_c2b, qe_n2g, qe_n2b, 128, LSEQ);
  conv_ln_kernel<<<LQ / 128, 256, smem128, stream>>>(
      q_a, q_b, q32, qe_c3w, qe_c3b, qe_n3g, qe_n3b, 128, LSEQ);      // q

  // ---- Values gather (transposed for TDM-friendly B operand) ----
  embed_t_kernel<<<(NSUP * HD + 255) / 256, 256, 0, stream>>>(
      sup_out, out_emb, valsT, NSUP, HD);

  // ---- Flash attention (partitioned) + combine ----
  flash_attn_kernel<<<dim3(LQ / 64, NPART), 256, smemfa, stream>>>(
      q_b, s_b, valsT, pacc, pm, pl, NSUP / NPART);
  attn_combine_kernel<<<LQ, 128, 0, stream>>>(pacc, pm, pl, att);

  // ---- Decoder ----
  decoder_kernel<<<LQ / 128, 256, smemdec, stream>>>(
      q32, att, dec_w1, dec_b1, dec_lng, dec_lnb, dec_w2, dec_b2, (float*)d_out);
}
